// Mamba_Layer_2302102470697
// MI455X (gfx1250) — compile-verified, matrix operand order checked by reading, not run on hardware
//
#include <hip/hip_runtime.h>
#include <hip/hip_bf16.h>
#include <math.h>

// Problem constants (match the reference)
#define B_      4
#define L_      4096
#define DMODEL  512
#define DIN     1024
#define DSTATE  16
#define DCONV   4
#define DTRANK  32
#define MT      (B_ * L_)   // 16384 token rows

typedef __attribute__((ext_vector_type(16))) __bf16 v16bf;
typedef __attribute__((ext_vector_type(8)))  __bf16 v8bf;
typedef __attribute__((ext_vector_type(8)))  float  v8f;

// ---------------------------------------------------------------------------
// Fragment loader (bf16 source): per CDNA5 ISA 16-bit A/B layout for
// 16x16x32 WMMA.  Lane l<16 holds row r=l, K = {k0..k0+7, k0+16..k0+23}
// with k0=0; lane l>=16 holds row r=l-16 with k0=8.  Two b128 loads, no cvt.
// ---------------------------------------------------------------------------
__device__ __forceinline__ v16bf load_frag(const __bf16* __restrict__ p) {
  v8bf lo = *(const v8bf*)p;          // K = k0 .. k0+7
  v8bf hi = *(const v8bf*)(p + 16);   // K = k0+16 .. k0+23
  return __builtin_shufflevector(lo, hi, 0, 1, 2, 3, 4, 5, 6, 7,
                                          8, 9, 10, 11, 12, 13, 14, 15);
}

// ---------------------------------------------------------------------------
// Generic C[M,N] = X[M,K] @ W[N,K]^T  via v_wmma_f32_16x16x32_bf16.
// A and B are pre-converted bf16.  One wave computes a 32x64 tile:
// 2 A fragments x 4 B fragments -> 8 accumulators; every B fragment feeds
// two WMMAs (24 fragment-bytes per WMMA vs 40 for a 16x64 tile).
// epilogue: 0 = none, 1 = softplus(val + bias[n])  (dt projection)
// C16 (optional): bf16 shadow copy of the output (feeds a later GEMM's A).
// ---------------------------------------------------------------------------
__global__ void __launch_bounds__(256)
wmma_gemm_xwT(const __bf16* __restrict__ X, int lda,
              const __bf16* __restrict__ W, int ldw,
              float* __restrict__ C, int ldc,
              int M, int N, int K,
              const float* __restrict__ bias, int epilogue,
              __bf16* __restrict__ C16)
{
  const int lane = threadIdx.x & 31;
  const int wave = threadIdx.x >> 5;
  const int m0 = (blockIdx.y * 8 + wave) * 32;   // 32 rows per wave
  const int n0 = blockIdx.x * 64;                // 64 cols per wave
  if (m0 >= M || n0 >= N) return;

  const int r16 = lane & 15;          // row (A) / column (B) index
  const int k0  = (lane >> 4) << 3;   // 0 or 8: K sub-offset per lane half

  const __bf16* aRow0 = X + (size_t)(m0 + r16) * lda + k0;
  const __bf16* aRow1 = aRow0 + (size_t)16 * (size_t)lda;
  const __bf16* bRow0 = W + (size_t)(n0 + r16) * ldw + k0;
  const size_t bstep = (size_t)16 * (size_t)ldw;

  v8f acc[8] = {};   // acc[j] = (m0..m0+15, nj) ; acc[4+j] = (m0+16.., nj)

  for (int k = 0; k < K; k += 32) {
    v16bf a0 = load_frag(aRow0 + k);
    v16bf a1 = load_frag(aRow1 + k);
    if (k + 32 < K) {
      __builtin_prefetch(aRow0 + k + 32, 0, 1);   // global_prefetch_b8
      __builtin_prefetch(aRow1 + k + 32, 0, 1);
      __builtin_prefetch(bRow0 + k + 32, 0, 1);
    }
#pragma unroll
    for (int j = 0; j < 4; ++j) {
      v16bf bf = load_frag(bRow0 + (size_t)j * bstep + k);
      acc[j] = __builtin_amdgcn_wmma_f32_16x16x32_bf16(
          false, a0, false, bf, (short)0, acc[j], false, false);
      acc[4 + j] = __builtin_amdgcn_wmma_f32_16x16x32_bf16(
          false, a1, false, bf, (short)0, acc[4 + j], false, false);
    }
  }

  // C/D layout: VGPR v -> M = mb + v ; lane&15 -> N
  const int nn = lane & 15;
  const int mb = (lane >> 4) << 3;
#pragma unroll
  for (int half = 0; half < 2; ++half) {
    const int mrow = m0 + half * 16 + mb;
#pragma unroll
    for (int j = 0; j < 4; ++j) {
      const int ncol = n0 + j * 16 + nn;
      float* cp = C + (size_t)mrow * ldc + ncol;
#pragma unroll
      for (int v = 0; v < 8; ++v) {
        float val = acc[half * 4 + j][v];
        if (epilogue == 1) {
          val += bias[ncol];
          val = (val > 20.f) ? val : log1pf(__expf(val));   // softplus
        }
        cp[(size_t)v * (size_t)ldc] = val;
        if (C16)
          C16[(size_t)(mrow + v) * (size_t)ldc + ncol] = (__bf16)val;
      }
    }
  }
}

// ---------------------------------------------------------------------------
// Bulk f32 -> bf16 conversion (8 elements / thread, b128-in, b128-out).
// ---------------------------------------------------------------------------
__global__ void __launch_bounds__(256)
cvt_f32_bf16(const float* __restrict__ src, __bf16* __restrict__ dst, int n)
{
  const int i = (blockIdx.x * 256 + threadIdx.x) * 8;
  if (i >= n) return;
  v8bf o;
#pragma unroll
  for (int j = 0; j < 8; ++j) o[j] = (__bf16)src[i + j];
  *(v8bf*)(dst + i) = o;
}

// ---------------------------------------------------------------------------
// Causal depthwise conv (4 taps) over time + SiLU.  Reads the xi half of xz,
// writes both f32 (scan input) and bf16 (x_proj GEMM A operand).
// ---------------------------------------------------------------------------
__global__ void __launch_bounds__(256)
conv_silu_kernel(const float* __restrict__ xz,
                 float* __restrict__ xi, __bf16* __restrict__ xi16,
                 const float* __restrict__ wconv, const float* __restrict__ bconv)
{
  const int idx = blockIdx.x * 256 + threadIdx.x;          // b*L*DIN elems
  const int d = idx & (DIN - 1);
  const int l = (idx >> 10) & (L_ - 1);
  const int b = idx >> 22;
  float s = bconv[d];
#pragma unroll
  for (int j = 0; j < DCONV; ++j) {
    const int ll = l - (DCONV - 1) + j;
    if (ll >= 0)
      s += xz[((size_t)(b * L_ + ll)) * (2 * DIN) + d] * wconv[d * DCONV + j];
  }
  const float v = s / (1.f + __expf(-s));                  // SiLU
  xi[idx]   = v;
  xi16[idx] = (__bf16)v;
}

// ---------------------------------------------------------------------------
// Selective scan.  One thread per (b, d, s) state element; h in one VGPR.
// y = sum_s h_s * C_s via width-16 __shfl_xor reduction; lane s==0 fuses the
// D-skip + SiLU(z) gate and writes the gated output as bf16 (out_proj A).
// ---------------------------------------------------------------------------
__global__ void __launch_bounds__(256)
scan_kernel(const float* __restrict__ xz, const float* __restrict__ xi,
            const float* __restrict__ xdbl, const float* __restrict__ dtbuf,
            const float* __restrict__ A_log, const float* __restrict__ Dvec,
            __bf16* __restrict__ yg16)
{
  const int t = blockIdx.x * 256 + threadIdx.x;   // 0 .. B*DIN*DSTATE-1
  const int s = t & 15;
  const int d = (t >> 4) & (DIN - 1);
  const int b = t >> 14;

  const float A  = -__expf(A_log[d * DSTATE + s]);
  const float Dd = Dvec[d];
  float h = 0.f;

  const size_t base = (size_t)b * L_;
  for (int l = 0; l < L_; ++l) {
    const size_t row = base + (size_t)l;
    const float dtv = dtbuf[row * DIN + d];
    const float xv  = xi[row * DIN + d];
    const float Bs  = xdbl[row * 64 + DTRANK + s];
    const float Cs  = xdbl[row * 64 + DTRANK + DSTATE + s];

    h = __expf(dtv * A) * h + (dtv * xv) * Bs;
    float p = h * Cs;
    p += __shfl_xor(p, 8, 16);
    p += __shfl_xor(p, 4, 16);
    p += __shfl_xor(p, 2, 16);
    p += __shfl_xor(p, 1, 16);

    if (s == 0) {
      const float zv = xz[row * (2 * DIN) + DIN + d];
      const float y  = p + xv * Dd;
      yg16[row * DIN + d] = (__bf16)(y * (zv / (1.f + __expf(-zv))));
    }
  }
}

// ---------------------------------------------------------------------------
// Residual add + LayerNorm over DMODEL=512; one 256-thread block per token.
// ---------------------------------------------------------------------------
__global__ void __launch_bounds__(256)
resid_ln_kernel(const float* __restrict__ x, const float* __restrict__ yproj,
                const float* __restrict__ g, const float* __restrict__ beta,
                float* __restrict__ out)
{
  const int rowi = blockIdx.x;
  const int t = threadIdx.x;
  const size_t off = (size_t)rowi * DMODEL;
  const float v0 = x[off + t]       + yproj[off + t];
  const float v1 = x[off + t + 256] + yproj[off + t + 256];

  __shared__ float s1[256], s2[256];
  s1[t] = v0 + v1;
  s2[t] = v0 * v0 + v1 * v1;
  __syncthreads();
  for (int o = 128; o > 0; o >>= 1) {
    if (t < o) { s1[t] += s1[t + o]; s2[t] += s2[t + o]; }
    __syncthreads();
  }
  const float mu  = s1[0] * (1.f / DMODEL);
  const float var = s2[0] * (1.f / DMODEL) - mu * mu;
  const float rs  = rsqrtf(var + 1e-5f);
  out[off + t]       = (v0 - mu) * rs * g[t]       + beta[t];
  out[off + t + 256] = (v1 - mu) * rs * g[t + 256] + beta[t + 256];
}

// ---------------------------------------------------------------------------
// Host-side orchestration
// ---------------------------------------------------------------------------
extern "C" void kernel_launch(void* const* d_in, const int* in_sizes, int n_in,
                              void* d_out, int out_size, void* d_ws, size_t ws_size,
                              hipStream_t stream) {
  const float* x       = (const float*)d_in[0];
  const float* w_in    = (const float*)d_in[1];
  const float* w_conv  = (const float*)d_in[2];
  const float* b_conv  = (const float*)d_in[3];
  const float* w_xproj = (const float*)d_in[4];
  const float* w_dt    = (const float*)d_in[5];
  const float* b_dt    = (const float*)d_in[6];
  const float* A_log   = (const float*)d_in[7];
  const float* Dv      = (const float*)d_in[8];
  const float* w_out   = (const float*)d_in[9];
  const float* ln_g    = (const float*)d_in[10];
  const float* ln_b    = (const float*)d_in[11];
  float* out = (float*)d_out;

  // ---- Workspace partition ----
  // f32 region
  float* ws    = (float*)d_ws;
  float* xz    = ws;                              // MT * 2048
  float* xi    = xz   + (size_t)MT * 2 * DIN;     // MT * 1024
  float* xdbl  = xi   + (size_t)MT * DIN;         // MT * 64
  float* dt    = xdbl + (size_t)MT * 64;          // MT * 1024
  float* yproj = dt   + (size_t)MT * DIN;         // MT * 512
  // bf16 region (16B-aligned: all preceding sizes are multiples of 4 floats)
  __bf16* bfr     = (__bf16*)(yproj + (size_t)MT * DMODEL);
  __bf16* xbf     = bfr;                              // MT * 512
  __bf16* w_in16  = xbf    + (size_t)MT * DMODEL;     // 2048 * 512
  __bf16* wx16    = w_in16 + (size_t)2 * DIN * DMODEL;// 64 * 1024
  __bf16* wdt16   = wx16   + (size_t)64 * DIN;        // 1024 * 32
  __bf16* wout16  = wdt16  + (size_t)DIN * DTRANK;    // 512 * 1024
  __bf16* xi16    = wout16 + (size_t)DMODEL * DIN;    // MT * 1024
  __bf16* xdbl16  = xi16   + (size_t)MT * DIN;        // MT * 64
  __bf16* yg16    = xdbl16 + (size_t)MT * 64;         // MT * 1024

  dim3 blk(256);

  // 0) one-pass f32 -> bf16 conversions (GEMM operands)
  cvt_f32_bf16<<<(MT * DMODEL) / 2048, blk, 0, stream>>>(x, xbf, MT * DMODEL);
  cvt_f32_bf16<<<(2 * DIN * DMODEL) / 2048, blk, 0, stream>>>(w_in, w_in16, 2 * DIN * DMODEL);
  cvt_f32_bf16<<<(64 * DIN) / 2048, blk, 0, stream>>>(w_xproj, wx16, 64 * DIN);
  cvt_f32_bf16<<<(DIN * DTRANK) / 2048, blk, 0, stream>>>(w_dt, wdt16, DIN * DTRANK);
  cvt_f32_bf16<<<(DMODEL * DIN) / 2048, blk, 0, stream>>>(w_out, wout16, DMODEL * DIN);

  // 1) in_proj: xz = x @ w_in^T          (M=16384, N=2048, K=512)
  wmma_gemm_xwT<<<dim3((2 * DIN) / 64, MT / 256), blk, 0, stream>>>(
      xbf, DMODEL, w_in16, DMODEL, xz, 2 * DIN, MT, 2 * DIN, DMODEL,
      nullptr, 0, nullptr);

  // 2) causal depthwise conv + SiLU -> xi (f32) + xi16 (bf16)
  conv_silu_kernel<<<(MT * DIN) / 256, blk, 0, stream>>>(xz, xi, xi16, w_conv, b_conv);

  // 3) x_proj: xdbl = xi @ w_xproj^T     (N=64, K=1024); bf16 shadow for dt GEMM
  wmma_gemm_xwT<<<dim3(64 / 64, MT / 256), blk, 0, stream>>>(
      xi16, DIN, wx16, DIN, xdbl, 64, MT, 64, DIN, nullptr, 0, xdbl16);

  // 4) dt = softplus(xdbl[:, :32] @ w_dt^T + b_dt)   (N=1024, K=32)
  wmma_gemm_xwT<<<dim3(DIN / 64, MT / 256), blk, 0, stream>>>(
      xdbl16, 64, wdt16, DTRANK, dt, DIN, MT, DIN, DTRANK, b_dt, 1, nullptr);

  // 5) selective scan + D-skip + SiLU(z) gate -> bf16 gated y
  scan_kernel<<<(B_ * DIN * DSTATE) / 256, blk, 0, stream>>>(
      xz, xi, xdbl, dt, A_log, Dv, yg16);

  // 6) out_proj: yproj = y_gated @ w_out^T   (N=512, K=1024)
  wmma_gemm_xwT<<<dim3(DMODEL / 64, MT / 256), blk, 0, stream>>>(
      yg16, DIN, wout16, DIN, yproj, DMODEL, MT, DMODEL, DIN,
      nullptr, 0, nullptr);

  // 7) residual + LayerNorm
  resid_ln_kernel<<<MT, blk, 0, stream>>>(x, yproj, ln_g, ln_b, out);
}
